// RNN_53755810677415
// MI455X (gfx1250) — matrix operand order repaired and checked by reading, not compiled, not run on hardware
//
#include <hip/hip_runtime.h>

// ---------------------------------------------------------------------------
// RNN forward for MI455X (gfx1250, wave32, WMMA + TDM).
//   x_gates = x @ W_xh^T + (b_xh + b_hh)        (bf16 WMMA GEMM)
//   for t:  h = tanh(x_gates[t] + h @ W_hh^T)   (one launch per step, WMMA)
//   out    = h @ W_dec^T + b_dec                (bf16 WMMA GEMM)
// A-tiles (shared by the 4 waves of a block) are staged into LDS with the
// Tensor Data Mover (tensor_load_to_lds + s_wait_tensorcnt), then fed to
// WMMA A-fragments via ds_load_b128.
// ---------------------------------------------------------------------------

typedef __attribute__((ext_vector_type(16))) __bf16 v16bf;
typedef __attribute__((ext_vector_type(8)))  __bf16 v8bf;
typedef __attribute__((ext_vector_type(8)))  float  v8f;
typedef __attribute__((ext_vector_type(4)))  unsigned int v4u;
typedef __attribute__((ext_vector_type(4)))  int    v4i;
typedef __attribute__((ext_vector_type(8)))  int    v8i;

#define B_   64
#define T_   512
#define DIN_ 256
#define H_   512
#define OUT_ 10000

// ---- workspace layout (bytes, 256-aligned) --------------------------------
static constexpr size_t OFF_XBF  = 0;                                   // [B*T, DIN] bf16
static constexpr size_t OFF_WXH  = OFF_XBF  + (size_t)B_*T_*DIN_*2;     // [H, DIN]  bf16
static constexpr size_t OFF_WHH  = OFF_WXH  + (size_t)H_*DIN_*2;        // [H, H]    bf16
static constexpr size_t OFF_WDEC = OFF_WHH  + (size_t)H_*H_*2;          // [OUT, H]  bf16
static constexpr size_t OFF_XG   = OFF_WDEC + (size_t)OUT_*H_*2;        // [T, B, H] f32
static constexpr size_t OFF_HF32 = OFF_XG   + (size_t)T_*B_*H_*4;       // [B, H]    f32
static constexpr size_t OFF_HBF0 = OFF_HF32 + (size_t)B_*H_*4;          // [B, H]    bf16
static constexpr size_t OFF_HBF1 = OFF_HBF0 + (size_t)B_*H_*2;          // [B, H]    bf16

// ---------------------------------------------------------------------------
// fast tanh (gfx1250 has V_TANH_F32)
// ---------------------------------------------------------------------------
__device__ __forceinline__ float fast_tanh(float x) {
#if __has_builtin(__builtin_amdgcn_tanhf)
    return __builtin_amdgcn_tanhf(x);
#else
    return tanhf(x);
#endif
}

// ---------------------------------------------------------------------------
// f32 -> bf16 conversion (grid-stride, one pass)
// ---------------------------------------------------------------------------
__global__ void cvt_f32_bf16(const float* __restrict__ in,
                             __bf16* __restrict__ out, int n) {
    int i = blockIdx.x * blockDim.x + threadIdx.x;
    int stride = gridDim.x * blockDim.x;
    for (; i < n; i += stride) out[i] = (__bf16)in[i];
}

// ---------------------------------------------------------------------------
// TDM: stage a contiguous [rows x rowlen] bf16 tile from global into LDS.
// Issued by the calling wave only (call from wave 0, then barrier).
// D# layout per CDNA5 ISA ch.8: group0 = {count, lds_addr, global_addr, type=2},
// group1 = {data_size=1(2B), tensor_dim0/1, tile_dim0/1, dim0_stride}.
// ---------------------------------------------------------------------------
__device__ __forceinline__ void tdm_stage_bf16(__bf16* ldst,
                                               const __bf16* gsrc,
                                               unsigned rows, unsigned rowlen,
                                               unsigned stride_elems) {
#if __has_builtin(__builtin_amdgcn_tensor_load_to_lds)
    // low 32 bits of the flat shared-aperture address == LDS byte offset
    unsigned lds_off_bytes = (unsigned)(size_t)ldst;
    unsigned long long ga = (unsigned long long)(size_t)gsrc;
    v4u g0;
    g0[0] = 1u;                                                  // count=1, user desc
    g0[1] = lds_off_bytes;                                       // D#.lds_addr
    g0[2] = (unsigned)ga;                                        // global_addr[31:0]
    g0[3] = (unsigned)((ga >> 32) & 0x01FFFFFFu) | (2u << 30);   // [56:32] | type=2
    v8i g1;
    g1[0] = (int)(1u << 16);                                     // wg_mask=0, data_size=2B
    g1[1] = (int)((rowlen & 0xFFFFu) << 16);                     // tensor_dim0[15:0]
    g1[2] = (int)((rowlen >> 16) | ((rows & 0xFFFFu) << 16));    // dim0[31:16]|dim1[15:0]
    g1[3] = (int)((rows >> 16) | ((rowlen & 0xFFFFu) << 16));    // dim1[31:16]|tile_dim0
    g1[4] = (int)(rows & 0xFFFFu);                               // tile_dim1 | tile_dim2=0
    g1[5] = (int)stride_elems;                                   // tensor_dim0_stride lo
    g1[6] = 0;                                                   // stride hi | dim1_stride
    g1[7] = 0;
    v4i gz = {};
#if __has_include(<hip/amd_detail/amd_gfx1250_TDM.h>)
    v8i gz8 = {};
    __builtin_amdgcn_tensor_load_to_lds(g0, g1, gz, gz, gz8, 0); // 6-arg toolchain
#else
    __builtin_amdgcn_tensor_load_to_lds(g0, g1, gz, gz, 0);      // 5-arg toolchain
#endif
    __builtin_amdgcn_s_wait_tensorcnt(0);
#else
    // Fallback (also parsed by the host pass): cooperative vector copy by
    // this wave; all staged regions are contiguous (stride == rowlen).
    (void)stride_elems;
    const unsigned lane = (unsigned)(threadIdx.x & 31);
    const unsigned total = rows * rowlen;
    for (unsigned i = lane * 8; i < total; i += 32 * 8)
        *(v8bf*)(ldst + i) = *(const v8bf*)(gsrc + i);
#endif
}

// ---------------------------------------------------------------------------
// One 16x16 f32 tile of A @ W^T via v_wmma_f32_16x16x32_bf16.
//   C[m,n] = sum_k A[arow+m, k] * W[ncol+n, k]
// A may point into LDS (staged tile) -> ds_load_b128 fragments.
// A-frag per ISA layout: lane<16 -> K {0..7,16..23}; lane>=16 -> K {8..15,24..31}
// B-frag: same K layout as A-frag, row (ncol + L%16) of W.
// ---------------------------------------------------------------------------
template <int K>
__device__ __forceinline__ v8f wmma_tile(const __bf16* A, int lda, int arow,
                                         const __bf16* __restrict__ W, int ldw, int ncol,
                                         int lane) {
    const int l16 = lane & 15;
    const int hi  = (lane >> 4) & 1;
    const __bf16* ap = A + (size_t)(arow + l16) * lda + hi * 8;
    const __bf16* bp = W + (size_t)(ncol + l16) * ldw + hi * 8;
    v8f acc = {};
#pragma unroll
    for (int kb = 0; kb < K; kb += 32) {
        v8bf a0 = *(const v8bf*)(ap + kb);
        v8bf a1 = *(const v8bf*)(ap + kb + 16);
        v8bf b0 = *(const v8bf*)(bp + kb);
        v8bf b1 = *(const v8bf*)(bp + kb + 16);
        v16bf a, b;
#pragma unroll
        for (int i = 0; i < 8; ++i) {
            a[i] = a0[i]; a[i + 8] = a1[i];
            b[i] = b0[i]; b[i + 8] = b1[i];
        }
        acc = __builtin_amdgcn_wmma_f32_16x16x32_bf16(
            /*neg_a=*/false, a, /*neg_b=*/false, b,
            /*c_mod=*/(short)0, acc, /*reuse_a=*/false, /*reuse_b=*/false);
    }
    return acc;
}

// ---------------------------------------------------------------------------
// Kernel 1: x_gates[t,b,n] = sum_k x[b,t,k]*W_xh[n,k] + b_xh[n] + b_hh[n]
// M = B*T = 32768 (2048 tiles), N = H = 512 (32 tiles), K = DIN = 256.
// 4 waves/block share one M-tile (16 rows of x) staged via TDM into LDS.
// ---------------------------------------------------------------------------
__global__ __launch_bounds__(128)
void xgates_kernel(const __bf16* __restrict__ xbf,
                   const __bf16* __restrict__ wxh,
                   const float* __restrict__ b_xh,
                   const float* __restrict__ b_hh,
                   float* __restrict__ xg) {
    __shared__ alignas(16) __bf16 asm_[16 * DIN_];     // 8 KB
    const int lane = threadIdx.x & 31;
    const int wave = threadIdx.x >> 5;
    const int tile = blockIdx.x * 4 + wave;            // 65536 tiles
    const int mt = tile >> 5;                          // same for all 4 waves
    const int nt = tile & 31;

    if (threadIdx.x < 32)
        tdm_stage_bf16(asm_, xbf + (size_t)mt * 16 * DIN_, 16, DIN_, DIN_);
    __syncthreads();

    v8f acc = wmma_tile<DIN_>(asm_, DIN_, 0, wxh, DIN_, nt * 16, lane);

    const int l16 = lane & 15;
    const int mo  = (lane >= 16) ? 8 : 0;
    const int n   = nt * 16 + l16;
    const float bias = b_xh[n] + b_hh[n];
#pragma unroll
    for (int v = 0; v < 8; ++v) {
        int m = mt * 16 + mo + v;                      // m = b*T + t
        int b = m >> 9;                                // / T_ (512)
        int t = m & (T_ - 1);
        xg[((size_t)t * B_ + b) * H_ + n] = acc[v] + bias;
    }
}

// ---------------------------------------------------------------------------
// Kernel 2: one recurrence step.  h_new = tanh(xg_t + h @ W_hh^T)
// M = 64 (4 tiles), N = 512 (32 tiles), K = 512.  32 blocks x 4 waves.
// The block's 16 h-rows are TDM-staged into LDS (shared by its 4 waves).
// ---------------------------------------------------------------------------
__global__ __launch_bounds__(128)
void rnn_step_kernel(const float* __restrict__ xg_t,
                     const __bf16* __restrict__ hin,
                     const __bf16* __restrict__ whh,
                     __bf16* __restrict__ hout,
                     float* __restrict__ hf32) {
    __shared__ alignas(16) __bf16 hsm[16 * H_];        // 16 KB
    const int lane = threadIdx.x & 31;
    const int wave = threadIdx.x >> 5;
    const int tile = blockIdx.x * 4 + wave;            // 128 tiles
    const int mt = tile >> 5;                          // same for all 4 waves
    const int nt = tile & 31;

    if (threadIdx.x < 32)
        tdm_stage_bf16(hsm, hin + (size_t)mt * 16 * H_, 16, H_, H_);
    __syncthreads();

    v8f acc = wmma_tile<H_>(hsm, H_, 0, whh, H_, nt * 16, lane);

    const int l16 = lane & 15;
    const int mo  = (lane >= 16) ? 8 : 0;
    const int n   = nt * 16 + l16;
#pragma unroll
    for (int v = 0; v < 8; ++v) {
        int m = mt * 16 + mo + v;                      // batch row 0..63
        float val = fast_tanh(acc[v] + xg_t[(size_t)m * H_ + n]);
        hf32[(size_t)m * H_ + n] = val;
        hout[(size_t)m * H_ + n] = (__bf16)val;
    }
}

// ---------------------------------------------------------------------------
// Kernel 3: decoder.  out[b,n] = sum_k h[b,k]*W_dec[n,k] + b_dec[n]
// M = 64 (4 tiles = 4 waves/block), N = 10000 (625 blocks), K = 512.
// Full h (64 x 512 bf16 = 64 KB) staged into LDS once per block via TDM.
// ---------------------------------------------------------------------------
__global__ __launch_bounds__(128)
void decoder_kernel(const __bf16* __restrict__ hbf,
                    const __bf16* __restrict__ wdec,
                    const float* __restrict__ b_dec,
                    float* __restrict__ out) {
    __shared__ alignas(16) __bf16 hsm[B_ * H_];        // 64 KB
    const int lane = threadIdx.x & 31;
    const int mt = threadIdx.x >> 5;                   // 0..3
    const int nt = blockIdx.x;                         // 0..624

    if (threadIdx.x < 32)
        tdm_stage_bf16(hsm, hbf, B_, H_, H_);
    __syncthreads();

    v8f acc = wmma_tile<H_>(hsm, H_, mt * 16, wdec, H_, nt * 16, lane);

    const int l16 = lane & 15;
    const int mo  = (lane >= 16) ? 8 : 0;
    const int n   = nt * 16 + l16;
    const float bias = b_dec[n];
#pragma unroll
    for (int v = 0; v < 8; ++v) {
        int m = mt * 16 + mo + v;
        out[(size_t)m * OUT_ + n] = acc[v] + bias;
    }
}

// ---------------------------------------------------------------------------
// Host launcher (graph-capture safe: only kernel launches + hipMemcpyAsync).
// ---------------------------------------------------------------------------
extern "C" void kernel_launch(void* const* d_in, const int* in_sizes, int n_in,
                              void* d_out, int out_size, void* d_ws, size_t ws_size,
                              hipStream_t stream) {
    const float* x     = (const float*)d_in[0];   // [B,T,DIN]
    const float* h0    = (const float*)d_in[1];   // [B,H]
    const float* W_xh  = (const float*)d_in[2];   // [H,DIN]
    const float* W_hh  = (const float*)d_in[3];   // [H,H]
    const float* b_xh  = (const float*)d_in[4];   // [H]
    const float* b_hh  = (const float*)d_in[5];   // [H]
    const float* W_dec = (const float*)d_in[6];   // [OUT,H]
    const float* b_dec = (const float*)d_in[7];   // [OUT]
    float* out = (float*)d_out;                   // [B*OUT] ++ [B*H]

    char* ws = (char*)d_ws;
    __bf16* xbf    = (__bf16*)(ws + OFF_XBF);
    __bf16* wxhbf  = (__bf16*)(ws + OFF_WXH);
    __bf16* whhbf  = (__bf16*)(ws + OFF_WHH);
    __bf16* wdecbf = (__bf16*)(ws + OFF_WDEC);
    float*  xg     = (float*)(ws + OFF_XG);
    float*  hf32   = (float*)(ws + OFF_HF32);
    __bf16* hbf[2] = { (__bf16*)(ws + OFF_HBF0), (__bf16*)(ws + OFF_HBF1) };

    // --- one-pass f32 -> bf16 conversions (L2-resident afterwards) ---------
    cvt_f32_bf16<<<1024, 256, 0, stream>>>(x,     xbf,    B_ * T_ * DIN_);
    cvt_f32_bf16<<<64,   256, 0, stream>>>(W_xh,  wxhbf,  H_ * DIN_);
    cvt_f32_bf16<<<128,  256, 0, stream>>>(W_hh,  whhbf,  H_ * H_);
    cvt_f32_bf16<<<1024, 256, 0, stream>>>(W_dec, wdecbf, OUT_ * H_);
    cvt_f32_bf16<<<32,   256, 0, stream>>>(h0,    hbf[0], B_ * H_);

    // --- big input GEMM: x_gates[T,B,H] ------------------------------------
    xgates_kernel<<<(2048 * 32) / 4, 128, 0, stream>>>(xbf, wxhbf, b_xh, b_hh, xg);

    // --- sequential recurrence: one launch per step (grid-wide sync) -------
    for (int t = 0; t < T_; ++t) {
        rnn_step_kernel<<<32, 128, 0, stream>>>(
            xg + (size_t)t * B_ * H_, hbf[t & 1], whhbf, hbf[(t + 1) & 1], hf32);
    }
    // after t = 511: bf16 h_last lives in hbf[0], f32 h_last in hf32

    // --- decoder on h_last --------------------------------------------------
    decoder_kernel<<<OUT_ / 16, 128, 0, stream>>>(hbf[0], wdecbf, b_dec, out);

    // --- second tuple output: h_last (f32) ---------------------------------
    (void)hipMemcpyAsync(out + (size_t)B_ * OUT_, hf32,
                         (size_t)B_ * H_ * sizeof(float),
                         hipMemcpyDeviceToDevice, stream);
}
